// AttentiveRNNLanguageModel_51402168599370
// MI455X (gfx1250) — matrix operand order repaired and checked by reading, not compiled, not run on hardware
//
#include <hip/hip_runtime.h>
#include <hip/hip_bf16.h>

// ---------------- problem constants ----------------
#define Vv   10000
#define Bb   32
#define Tt   512
#define Hh   512
#define Pp   20
#define G4H  (4*Hh)      // 2048
#define MROWS (Bb*Tt)    // 16384

typedef __attribute__((ext_vector_type(16))) __bf16 v16bf;
typedef __attribute__((ext_vector_type(8)))  float  v8f;
typedef unsigned short ushortt;

// ---------------- helpers ----------------
__device__ __forceinline__ ushortt f2bf(float f) {
    unsigned u = __builtin_bit_cast(unsigned, f);
    unsigned r = (u + 0x7FFFu + ((u >> 16) & 1u)) >> 16;
    return (ushortt)r;
}
__device__ __forceinline__ float sigmoidf_(float x) { return 1.0f / (1.0f + __expf(-x)); }

union FragU { uint4 u[2]; v16bf v; };

// A fragment (16x32 bf16, M x K): lane holds row M=lane&15; lanes<16 get K 0-7 & 16-23,
// lanes>=16 get K 8-15 & 24-31 (ISA 7.12.2 16-bit A layout).
__device__ __forceinline__ v16bf frag_a(const ushortt* As, int ld, int row0, int lane) {
    FragU f;
    int r  = row0 + (lane & 15);
    int kb = (lane >> 4) << 3;
    f.u[0] = *(const uint4*)&As[r * ld + kb];
    f.u[1] = *(const uint4*)&As[r * ld + kb + 16];
    return f.v;
}
// B fragment (32x16 bf16, K x N) from a row-major [N,K] LDS tile: lane holds col N=lane&15;
// lanes<16 K 0-7 & 16-23, lanes>=16 K 8-15 & 24-31.
__device__ __forceinline__ v16bf frag_b(const ushortt* Bs, int ld, int col0, int lane) {
    FragU f;
    int c  = col0 + (lane & 15);
    int kb = (lane >> 4) << 3;
    f.u[0] = *(const uint4*)&Bs[c * ld + kb];
    f.u[1] = *(const uint4*)&Bs[c * ld + kb + 16];
    return f.v;
}

// ---- CDNA5 async global->LDS (ASYNCcnt-tracked, bypasses VGPRs) ----
__device__ __forceinline__ void async_ld16(unsigned ldsoff, const ushortt* gptr) {
    asm volatile("global_load_async_to_lds_b128 %0, %1, off"
                 :: "v"(ldsoff), "v"(gptr) : "memory");
}
__device__ __forceinline__ void wait_async(int pending4) {
    if (pending4) asm volatile("s_wait_asynccnt 0x4" ::: "memory");
    else          asm volatile("s_wait_asynccnt 0x0" ::: "memory");
}
__device__ __forceinline__ unsigned lds_off(const void* p) {
    return (unsigned)(size_t)p;   // LDS aperture: addr[31:0] is the LDS byte offset
}

// ---------------- generic elementwise kernels ----------------
__global__ __launch_bounds__(256) void cvt_f32_bf16(const float* __restrict__ src,
                                                    ushortt* __restrict__ dst, int n) {
    int i = blockIdx.x * 256 + threadIdx.x;
    if (i < n) dst[i] = f2bf(src[i]);
}

__global__ __launch_bounds__(256) void embed_gather(const int* __restrict__ idx,
                                                    const float* __restrict__ table,
                                                    ushortt* __restrict__ out) {
    int i = blockIdx.x * 256 + threadIdx.x;           // over MROWS*Hh
    if (i >= MROWS * Hh) return;
    int bt = i >> 9;            // /Hh
    int h  = i & (Hh - 1);
    int v  = idx[bt];
    out[i] = f2bf(table[(size_t)v * Hh + h]);
}

// ---------------- WMMA GEMM: C[M,N] = act(A[M,K] @ Bt[N,K]^T + bias) ----------------
// Block tile 128(M) x 128(N), K-step 32, double-buffered LDS fed by async
// global->LDS loads. 256 threads = 8 wave32 in a 4x2 grid; each wave owns a
// 32x64 patch (2x4 wmma tiles -> 8 WMMAs per 6 fragment loads per K chunk).
#define LDA 40
#define LDB 40
template<int OUT_BF16, int ACT>
__global__ __launch_bounds__(256) void gemm_wmma(const ushortt* __restrict__ A,
                                                 const ushortt* __restrict__ Bt,
                                                 const float* __restrict__ bias,
                                                 float* __restrict__ Cf,
                                                 ushortt* __restrict__ Cb,
                                                 int M, int N, int K) {
    __shared__ ushortt AsBuf[2][128 * LDA];
    __shared__ ushortt BsBuf[2][128 * LDB];
    const int tid  = threadIdx.x;
    const int wave = tid >> 5, lane = tid & 31;
    const int wr = wave >> 1, wc = wave & 1;           // 4 x 2 wave grid
    const int n0 = blockIdx.x * 128;
    const int m0 = blockIdx.y * 128;

    v8f zero = {0.f,0.f,0.f,0.f,0.f,0.f,0.f,0.f};
    v8f acc[2][4] = {{zero, zero, zero, zero}, {zero, zero, zero, zero}};

    // staging assignment: 2 x b128 for A tile + 2 x b128 for B tile per thread
    const int sr0 = (tid * 2)     >> 2, sc0 = ((tid * 2)     & 3) << 3;
    const int sr1 = (tid * 2 + 1) >> 2, sc1 = ((tid * 2 + 1) & 3) << 3;

    const int KC = K >> 5;
    // prologue: chunk 0 -> buffer 0
    async_ld16(lds_off(&AsBuf[0][sr0 * LDA + sc0]), &A[(size_t)(m0 + sr0) * K + sc0]);
    async_ld16(lds_off(&AsBuf[0][sr1 * LDA + sc1]), &A[(size_t)(m0 + sr1) * K + sc1]);
    async_ld16(lds_off(&BsBuf[0][sr0 * LDB + sc0]), &Bt[(size_t)(n0 + sr0) * K + sc0]);
    async_ld16(lds_off(&BsBuf[0][sr1 * LDB + sc1]), &Bt[(size_t)(n0 + sr1) * K + sc1]);

    for (int kc = 0; kc < KC; ++kc) {
        const int cur = kc & 1;
        const int pending = (kc + 1 < KC);
        if (pending) {
            const int k1 = (kc + 1) << 5;
            async_ld16(lds_off(&AsBuf[cur ^ 1][sr0 * LDA + sc0]), &A[(size_t)(m0 + sr0) * K + k1 + sc0]);
            async_ld16(lds_off(&AsBuf[cur ^ 1][sr1 * LDA + sc1]), &A[(size_t)(m0 + sr1) * K + k1 + sc1]);
            async_ld16(lds_off(&BsBuf[cur ^ 1][sr0 * LDB + sc0]), &Bt[(size_t)(n0 + sr0) * K + k1 + sc0]);
            async_ld16(lds_off(&BsBuf[cur ^ 1][sr1 * LDB + sc1]), &Bt[(size_t)(n0 + sr1) * K + k1 + sc1]);
        }
        wait_async(pending);   // current chunk complete (in-flight: next chunk's 4)
        __syncthreads();

        const ushortt* As = AsBuf[cur];
        const ushortt* Bs = BsBuf[cur];
        v16bf a0 = frag_a(As, LDA, wr * 32,      lane);
        v16bf a1 = frag_a(As, LDA, wr * 32 + 16, lane);
        #pragma unroll
        for (int n = 0; n < 4; ++n) {
            v16bf b = frag_b(Bs, LDB, wc * 64 + n * 16, lane);
            acc[0][n] = __builtin_amdgcn_wmma_f32_16x16x32_bf16(false, a0, false, b, (short)0, acc[0][n], false, false);
            acc[1][n] = __builtin_amdgcn_wmma_f32_16x16x32_bf16(false, a1, false, b, (short)0, acc[1][n], false, false);
        }
        __syncthreads();       // everyone done reading `cur` before it is re-staged
    }

    const int hi8 = (lane >> 4) << 3;
    #pragma unroll
    for (int a = 0; a < 2; ++a)
    #pragma unroll
    for (int n = 0; n < 4; ++n) {
        int col = n0 + wc * 64 + n * 16 + (lane & 15);
        if (col >= N) continue;
        float bv = bias ? bias[col] : 0.0f;
        int rbase = m0 + wr * 32 + a * 16 + hi8;
        #pragma unroll
        for (int r = 0; r < 8; ++r) {
            float v = acc[a][n][r] + bv;
            if (ACT == 1) v = tanhf(v);
            size_t o = (size_t)(rbase + r) * N + col;
            if (OUT_BF16) Cb[o] = f2bf(v); else Cf[o] = v;
        }
    }
}

// ---------------- global spin barrier ----------------
__device__ __forceinline__ void grid_barrier(unsigned* cnt, volatile unsigned* gen,
                                             int nwg, unsigned phase) {
    __syncthreads();
    if (threadIdx.x == 0) {
        __threadfence();
        unsigned v = atomicAdd(cnt, 1u);
        if (v == (unsigned)(nwg - 1)) {
            *cnt = 0u;
            __threadfence();
            atomicAdd((unsigned*)gen, 1u);
        } else {
            while (*gen < phase) { __builtin_amdgcn_s_sleep(2); }
        }
    }
    __syncthreads();
}

// ---------------- encoder LSTM: W_hh slice resident in LDS ----------------
#define NWG_LSTM 16
#define HU (Hh / NWG_LSTM)     // 32 hidden units per WG; 4*HU = 128 gate cols
#define WLD 520                // 512 + 8 pad (halfs)
#define HLD 520
#define GLD 132
__global__ __launch_bounds__(256) void lstm_enc(const float* __restrict__ gx,
                                                const float* __restrict__ W_hh,
                                                const float* __restrict__ b_ih,
                                                const float* __restrict__ b_hh,
                                                float* __restrict__ enc,
                                                ushortt* __restrict__ enc_bf,
                                                ushortt* __restrict__ h_buf,
                                                unsigned* __restrict__ ctrl) {
    extern __shared__ char smem[];
    ushortt* whh_s = (ushortt*)smem;                     // [128][WLD]
    ushortt* hs    = whh_s + 128 * WLD;                  // [32][HLD]
    float*   gs    = (float*)(hs + 32 * HLD);            // [32][GLD]
    float*   cs    = gs + 32 * GLD;                      // [32][HU+1]

    const int tid  = threadIdx.x;
    const int wg   = blockIdx.x;
    const int wave = tid >> 5, lane = tid & 31;
    unsigned* cnt = ctrl;
    volatile unsigned* gen = ctrl + 16;

    // Load W_hh slice once (rows: 4 gates x HU units of this WG), f32 -> bf16 in LDS.
    for (int i = tid; i < 128 * Hh; i += 256) {
        int lr = i >> 9, c = i & (Hh - 1);
        int gate = lr >> 5, u = lr & (HU - 1);
        int grow = gate * Hh + wg * HU + u;
        whh_s[lr * WLD + c] = f2bf(W_hh[(size_t)grow * Hh + c]);
    }
    // c state and this WG's h_buf slice = 0
    for (int i = tid; i < Bb * HU; i += 256) {
        int b = i >> 5, u = i & (HU - 1);
        cs[b * (HU + 1) + u] = 0.0f;
        h_buf[b * Hh + wg * HU + u] = 0;
    }
    unsigned phase = 1;
    grid_barrier(cnt, gen, NWG_LSTM, phase++);

    for (int t = 0; t < Tt; ++t) {
        // pull full h (bf16, 32x512) into LDS
        for (int i = tid; i < (Bb * Hh) / 8; i += 256) {   // 2048 uint4 chunks
            int b = i >> 6, c = (i & 63) << 3;
            *(uint4*)&hs[b * HLD + c] = *(const uint4*)&h_buf[b * Hh + c];
        }
        __syncthreads();

        // gates[32 x 128] += h @ Whh_slice^T via WMMA; wave w owns col tile w
        v8f zero = {0.f,0.f,0.f,0.f,0.f,0.f,0.f,0.f};
        v8f acc[2] = {zero, zero};
        for (int kc = 0; kc < Hh / 32; ++kc) {
            v16bf vb = frag_b(whh_s + kc * 32, WLD, wave * 16, lane);
            v16bf va0 = frag_a(hs + kc * 32, HLD, 0,  lane);
            v16bf va1 = frag_a(hs + kc * 32, HLD, 16, lane);
            acc[0] = __builtin_amdgcn_wmma_f32_16x16x32_bf16(false, va0, false, vb, (short)0, acc[0], false, false);
            acc[1] = __builtin_amdgcn_wmma_f32_16x16x32_bf16(false, va1, false, vb, (short)0, acc[1], false, false);
        }
        const int hi8 = (lane >> 4) << 3;
        #pragma unroll
        for (int mi = 0; mi < 2; ++mi)
        #pragma unroll
        for (int r = 0; r < 8; ++r)
            gs[(mi * 16 + hi8 + r) * GLD + wave * 16 + (lane & 15)] = acc[mi][r];
        __syncthreads();

        // elementwise LSTM update for this WG's HU units
        for (int i = tid; i < Bb * HU; i += 256) {
            int b = i >> 5, u = i & (HU - 1);
            size_t gxb = ((size_t)(b * Tt + t)) * G4H;
            int c0 = 0 * Hh + wg * HU + u, c1 = 1 * Hh + wg * HU + u;
            int c2 = 2 * Hh + wg * HU + u, c3 = 3 * Hh + wg * HU + u;
            float gi = gs[b * GLD +  0 + u] + gx[gxb + c0] + b_ih[c0] + b_hh[c0];
            float gf = gs[b * GLD + 32 + u] + gx[gxb + c1] + b_ih[c1] + b_hh[c1];
            float gg = gs[b * GLD + 64 + u] + gx[gxb + c2] + b_ih[c2] + b_hh[c2];
            float go = gs[b * GLD + 96 + u] + gx[gxb + c3] + b_ih[c3] + b_hh[c3];
            float c = sigmoidf_(gf) * cs[b * (HU + 1) + u] + sigmoidf_(gi) * tanhf(gg);
            float h = sigmoidf_(go) * tanhf(c);
            cs[b * (HU + 1) + u] = c;
            size_t eo = ((size_t)(b * Tt + t)) * Hh + wg * HU + u;
            enc[eo] = h;
            ushortt hb = f2bf(h);
            enc_bf[eo] = hb;
            h_buf[b * Hh + wg * HU + u] = hb;
        }
        grid_barrier(cnt, gen, NWG_LSTM, phase++);
    }
}

// ---------------- positional LSTM (P=20) + mu/sig + mu recurrence, 1 WG ----------------
__global__ __launch_bounds__(640) void pos_lstm(const float* __restrict__ gpx,
                                                const float* __restrict__ Wp_hh,
                                                const float* __restrict__ bp_ih,
                                                const float* __restrict__ bp_hh,
                                                const float* __restrict__ W_mu,
                                                const float* __restrict__ b_mu,
                                                const float* __restrict__ W_sig,
                                                const float* __restrict__ b_sig,
                                                float* __restrict__ mu_g,
                                                float* __restrict__ sig_g) {
    __shared__ float hp[Bb][Pp], cp[Bb][Pp], wph[4 * Pp][Pp];
    __shared__ float wmu[3][Pp], wsg[Pp], muw[Bb][4], mu_prev[Bb];
    int tid = threadIdx.x;
    for (int i = tid; i < 4 * Pp * Pp; i += 640) wph[i / Pp][i % Pp] = Wp_hh[i];
    for (int i = tid; i < 3 * Pp; i += 640) wmu[i / Pp][i % Pp] = W_mu[i];
    if (tid < Pp) wsg[tid] = W_sig[tid];
    if (tid < Bb * Pp) { hp[tid / Pp][tid % Pp] = 0.0f; cp[tid / Pp][tid % Pp] = 0.0f; }
    if (tid < Bb) mu_prev[tid] = 0.0f;
    __syncthreads();
    int b = tid / Pp, p = tid % Pp;
    for (int t = 0; t < Tt; ++t) {
        const float* gx = &gpx[((size_t)(b * Tt + t)) * (4 * Pp)];
        float g[4];
        #pragma unroll
        for (int gt = 0; gt < 4; ++gt) {
            float a = gx[gt * Pp + p] + bp_ih[gt * Pp + p] + bp_hh[gt * Pp + p];
            for (int q = 0; q < Pp; ++q) a += hp[b][q] * wph[gt * Pp + p][q];
            g[gt] = a;
        }
        __syncthreads();
        float c = sigmoidf_(g[1]) * cp[b][p] + sigmoidf_(g[0]) * tanhf(g[2]);
        cp[b][p] = c;
        hp[b][p] = sigmoidf_(g[3]) * tanhf(c);
        __syncthreads();
        if (p < 3) {
            float a = b_mu[p];
            for (int q = 0; q < Pp; ++q) a += hp[b][q] * wmu[p][q];
            muw[b][p] = fmaxf(a, 0.0f);
        } else if (p == 3) {
            float a = b_sig[0];
            for (int q = 0; q < Pp; ++q) a += hp[b][q] * wsg[q];
            muw[b][3] = sigmoidf_(a);
        }
        __syncthreads();
        if (p == 0) {
            float m = muw[b][0] * mu_prev[b] + muw[b][1] * (1.0f / Tt)
                    + muw[b][2] * (t + 1.0f) / Tt;
            mu_prev[b] = m;
            mu_g[b * Tt + t] = m;
            sig_g[b * Tt + t] = muw[b][3];
        }
        __syncthreads();
    }
}

// ---------------- attention weights + context + cat pack ----------------
__global__ __launch_bounds__(256) void attn_ctx(const float* __restrict__ enc,
                                                const ushortt* __restrict__ enc_bf,
                                                const float* __restrict__ mu_g,
                                                const float* __restrict__ sig_g,
                                                ushortt* __restrict__ cat) {
    __shared__ float wbuf[Tt];
    __shared__ float rbuf[256];
    int tid = threadIdx.x;
    int j = blockIdx.x & (Tt - 1);
    int b = blockIdx.x >> 9;
    float mu = mu_g[b * Tt + j];
    float sg = sig_g[b * Tt + j];
    float inv = 1.0f / (2.0f * sg * sg);
    float ss = 0.0f;
    for (int t = tid; t < Tt; t += 256) {
        float v = 0.0f;
        if (t <= j) {
            float d = (float)t / (j + 1.0f) - mu;
            v = __expf(-d * d * inv);
        }
        wbuf[t] = v;
        ss += v * v;
    }
    rbuf[tid] = ss;
    __syncthreads();
    for (int s = 128; s > 0; s >>= 1) {
        if (tid < s) rbuf[tid] += rbuf[tid + s];
        __syncthreads();
    }
    float scale = 1.0f / fmaxf(sqrtf(rbuf[0]), 1e-12f);
    for (int t = tid; t < Tt; t += 256) wbuf[t] *= scale;
    __syncthreads();
    size_t row = (size_t)(b * Tt + j);
    for (int h = tid; h < Hh; h += 256) {
        float acc = 0.0f;
        for (int t = 0; t <= j; ++t) acc += wbuf[t] * enc[((size_t)(b * Tt + t)) * Hh + h];
        cat[row * (2 * Hh) + h]      = f2bf(acc);
        cat[row * (2 * Hh) + Hh + h] = enc_bf[row * Hh + h];
    }
}

// ---------------- launcher ----------------
extern "C" void kernel_launch(void* const* d_in, const int* in_sizes, int n_in,
                              void* d_out, int out_size, void* d_ws, size_t ws_size,
                              hipStream_t stream) {
    const int*   inp    = (const int*)  d_in[0];
    const float* emb_w  = (const float*)d_in[3];
    const float* dec_b  = (const float*)d_in[4];
    const float* W_ih   = (const float*)d_in[5];
    const float* W_hh   = (const float*)d_in[6];
    const float* b_ih   = (const float*)d_in[7];
    const float* b_hh   = (const float*)d_in[8];
    const float* Wp_ih  = (const float*)d_in[9];
    const float* Wp_hh  = (const float*)d_in[10];
    const float* bp_ih  = (const float*)d_in[11];
    const float* bp_hh  = (const float*)d_in[12];
    const float* W_mu   = (const float*)d_in[13];
    const float* b_mu   = (const float*)d_in[14];
    const float* W_sig  = (const float*)d_in[15];
    const float* b_sig  = (const float*)d_in[16];
    const float* W_cat  = (const float*)d_in[17];
    const float* b_cat  = (const float*)d_in[18];

    char* ws = (char*)d_ws;
    size_t off = 0;
    auto alloc = [&](size_t bytes) -> void* {
        void* p = ws + off;
        off = (off + bytes + 255) & ~(size_t)255;
        return p;
    };
    unsigned* ctrl      = (unsigned*)alloc(256);
    ushortt*  embT_bf   = (ushortt*) alloc((size_t)Vv * Hh * 2);
    ushortt*  emb_bf    = (ushortt*) alloc((size_t)MROWS * Hh * 2);
    ushortt*  wih_bf    = (ushortt*) alloc((size_t)G4H * Hh * 2);
    ushortt*  wpih_bf   = (ushortt*) alloc((size_t)4 * Pp * Hh * 2);
    ushortt*  wcat_bf   = (ushortt*) alloc((size_t)Hh * 2 * Hh * 2);
    float*    gx        = (float*)   alloc((size_t)MROWS * G4H * 4);
    ushortt*  h_buf     = (ushortt*) alloc((size_t)Bb * Hh * 2);
    float*    enc       = (float*)   alloc((size_t)MROWS * Hh * 4);
    ushortt*  enc_bf    = (ushortt*) alloc((size_t)MROWS * Hh * 2);
    float*    gpx       = (float*)   alloc((size_t)MROWS * 4 * Pp * 4);
    float*    mu_g      = (float*)   alloc((size_t)MROWS * 4);
    float*    sig_g     = (float*)   alloc((size_t)MROWS * 4);
    ushortt*  cat       = (ushortt*) alloc((size_t)MROWS * 2 * Hh * 2);
    ushortt*  comb      = (ushortt*) alloc((size_t)MROWS * Hh * 2);

    hipMemsetAsync(ctrl, 0, 256, stream);

    // weight + activation conversions to bf16
    cvt_f32_bf16<<<(Vv * Hh + 255) / 256, 256, 0, stream>>>(emb_w, embT_bf, Vv * Hh);
    cvt_f32_bf16<<<(G4H * Hh + 255) / 256, 256, 0, stream>>>(W_ih, wih_bf, G4H * Hh);
    cvt_f32_bf16<<<(4 * Pp * Hh + 255) / 256, 256, 0, stream>>>(Wp_ih, wpih_bf, 4 * Pp * Hh);
    cvt_f32_bf16<<<(Hh * 2 * Hh + 255) / 256, 256, 0, stream>>>(W_cat, wcat_bf, Hh * 2 * Hh);
    embed_gather<<<(MROWS * Hh + 255) / 256, 256, 0, stream>>>(inp, emb_w, emb_bf);

    // gx = emb @ W_ih^T  [16384 x 2048]
    gemm_wmma<0, 0><<<dim3((G4H + 127) / 128, MROWS / 128), 256, 0, stream>>>(
        emb_bf, wih_bf, nullptr, gx, nullptr, MROWS, G4H, Hh);

    // encoder LSTM (persistent, 16 WGs, W_hh slice resident in LDS)
    size_t lstm_lds = (size_t)(128 * WLD + 32 * HLD) * 2 + (size_t)(32 * GLD + 32 * (HU + 1)) * 4;
    lstm_enc<<<NWG_LSTM, 256, lstm_lds, stream>>>(gx, W_hh, b_ih, b_hh,
                                                  enc, enc_bf, h_buf, ctrl);

    // gpx = enc @ Wp_ih^T  [16384 x 80]
    gemm_wmma<0, 0><<<dim3(1, MROWS / 128), 256, 0, stream>>>(
        enc_bf, wpih_bf, nullptr, gpx, nullptr, MROWS, 4 * Pp, Hh);

    pos_lstm<<<1, 640, 0, stream>>>(gpx, Wp_hh, bp_ih, bp_hh, W_mu, b_mu,
                                    W_sig, b_sig, mu_g, sig_g);

    attn_ctx<<<Bb * Tt, 256, 0, stream>>>(enc, enc_bf, mu_g, sig_g, cat);

    // comb = tanh(cat @ W_cat^T + b_cat)  [16384 x 512], bf16 out
    gemm_wmma<1, 1><<<dim3((Hh + 127) / 128, MROWS / 128), 256, 0, stream>>>(
        cat, wcat_bf, b_cat, nullptr, comb, MROWS, Hh, 2 * Hh);

    // decoded = comb @ embedding^T + dec_bias  [16384 x 10000]
    gemm_wmma<0, 0><<<dim3((Vv + 127) / 128, MROWS / 128), 256, 0, stream>>>(
        comb, embT_bf, dec_b, (float*)d_out, nullptr, MROWS, Vv, Hh);
}
